// AttentionDecoder_25409026523668
// MI455X (gfx1250) — matrix operand order repaired and checked by reading, not compiled, not run on hardware
//
#include <hip/hip_runtime.h>

#define H_  512
#define D_  512
#define A_  256
#define V_  10000
#define TE_ 256
#define TD_ 128

typedef __attribute__((ext_vector_type(16))) __bf16 v16bf;
typedef __attribute__((ext_vector_type(8)))  __bf16 v8bf;
typedef __attribute__((ext_vector_type(8)))  float  v8f;

static __device__ __forceinline__ v8f wmma_bf16(v16bf a, v16bf b, v8f c) {
  return __builtin_amdgcn_wmma_f32_16x16x32_bf16(false, a, false, b, (short)0, c,
                                                 false, false);
}

// ---- fragment packing (wave32 layouts per CDNA5 ISA 7.12.2) -----------------
// A 16x32 bf16: lane L -> row m=L&15; elements 0..7 = K kb..kb+7, 8..15 = kb+16..kb+23
// where kb = k0 + (L>>4)*8.
static __device__ __forceinline__ v16bf pack_a(const float* p) {
  float t[16];
  *(float4*)(t + 0)  = ((const float4*)p)[0];
  *(float4*)(t + 4)  = ((const float4*)p)[1];
  *(float4*)(t + 8)  = ((const float4*)p)[4];
  *(float4*)(t + 12) = ((const float4*)p)[5];
  v16bf a;
#pragma unroll
  for (int e = 0; e < 16; ++e) a[e] = (__bf16)t[e];
  return a;
}
static __device__ __forceinline__ v16bf pack_a(const __bf16* p) {
  v8bf lo = *(const v8bf*)p;
  v8bf hi = *(const v8bf*)(p + 16);
  v16bf a;
#pragma unroll
  for (int e = 0; e < 8; ++e) { a[e] = lo[e]; a[e + 8] = hi[e]; }
  return a;
}
// B 32x16 bf16 (B = W^T, W row-major [N,K]): lane L -> col n=L&15,
// element e -> K = k0 + (L>>4)*8 + e (e<8), k0 + 16 + (L>>4)*8 + (e-8) (e>=8).
static __device__ __forceinline__ v16bf pack_b(const float* p) {
  float t[16];
  *(float4*)(t + 0)  = ((const float4*)p)[0];
  *(float4*)(t + 4)  = ((const float4*)p)[1];
  *(float4*)(t + 8)  = ((const float4*)p)[4];
  *(float4*)(t + 12) = ((const float4*)p)[5];
  v16bf b;
#pragma unroll
  for (int e = 0; e < 16; ++e) b[e] = (__bf16)t[e];
  return b;
}

// ---- generic C[M,N] = A[M,K] @ W[N,K]^T + bias, bf16 WMMA f32-accumulate ----
// block = 256 thr (8 waves); wave owns 128(M) x 32(N); grid.y covers M/128.
template<typename AT, typename OT>
__global__ void gemm_bt(const AT* __restrict__ A, const float* __restrict__ W,
                        const float* __restrict__ bias, OT* __restrict__ C,
                        int N, int K, int lda, int ldb, int ldc) {
  const int lane = threadIdx.x & 31;
  const int wv   = threadIdx.x >> 5;
  const int n0   = (blockIdx.x * 8 + wv) * 32;
  if (n0 >= N) return;
  const bool n1ok = (n0 + 32) <= N;
  const int m0   = blockIdx.y * 128;
  const int nA   = n0 + (lane & 15);
  const int nB   = n1ok ? (nA + 16) : nA;
  const int kbB  = (lane >> 4) << 3;
  const int kbA  = (lane >> 4) << 3;
  const int ml   = m0 + (lane & 15);

  const v8f vz = {0.f, 0.f, 0.f, 0.f, 0.f, 0.f, 0.f, 0.f};
  v8f acc[16];
#pragma unroll
  for (int i = 0; i < 16; ++i) acc[i] = vz;

#pragma unroll 2
  for (int k0 = 0; k0 < K; k0 += 32) {
    const float* wp0 = W + (size_t)nA * ldb + k0 + kbB;
    const float* wp1 = W + (size_t)nB * ldb + k0 + kbB;
    if (k0 + 512 < K) __builtin_prefetch(wp0 + 512, 0, 1);  // global_prefetch_b8
    v16bf b0 = pack_b(wp0);
    v16bf b1 = pack_b(wp1);
#pragma unroll
    for (int mt = 0; mt < 8; ++mt) {
      v16bf a = pack_a(A + (size_t)(ml + mt * 16) * lda + k0 + kbA);
      acc[2 * mt]     = wmma_bf16(a, b0, acc[2 * mt]);
      acc[2 * mt + 1] = wmma_bf16(a, b1, acc[2 * mt + 1]);
    }
  }
  const float bv0 = bias ? bias[nA] : 0.f;
  const float bv1 = (bias && n1ok) ? bias[nB] : 0.f;
  const int mr = m0 + ((lane >> 4) << 3);
#pragma unroll
  for (int mt = 0; mt < 8; ++mt) {
#pragma unroll
    for (int i = 0; i < 8; ++i) {
      const int m = mr + mt * 16 + i;
      C[(size_t)m * ldc + nA] = (OT)(acc[2 * mt][i] + bv0);
      if (n1ok) C[(size_t)m * ldc + nB] = (OT)(acc[2 * mt + 1][i] + bv1);
    }
  }
}

// ---- encoder gates: h = (1-z)*n with zero hidden (gh = bh_e only) -----------
__global__ void enc_gates_kernel(const float* __restrict__ GI, const float* __restrict__ bh,
                                 float* __restrict__ encH, float* __restrict__ hcur) {
  int idx = blockIdx.x * blockDim.x + threadIdx.x;
  if (idx >= TE_ * H_) return;
  int tt = idx >> 9;
  int j  = idx & (H_ - 1);
  const float* gi = GI + (size_t)tt * (3 * H_);
  float r = 1.f / (1.f + __expf(-(gi[j] + bh[j])));
  float z = 1.f / (1.f + __expf(-(gi[H_ + j] + bh[H_ + j])));
  float n = tanhf(gi[2 * H_ + j] + r * bh[2 * H_ + j]);
  float h = (1.f - z) * n;
  encH[idx] = h;
  if (tt == TE_ - 1) hcur[j] = h;
}

__global__ void init_bar_kernel(unsigned int* bar) { *bar = 0u; }

// ---- decoder GRU scan: 8 persistent WGs, Wh_d slice resident in LDS (bf16),
// matvec via WMMA (wave = one 16-row tile; B = h broadcast), global spin barrier.
// Two independent accumulator chains (even/odd K chunks) keep the matrix pipe fed.
__global__ void decoder_scan(const float* __restrict__ Whd, const float* __restrict__ bhd,
                             const float* __restrict__ GI, const float* __restrict__ h0,
                             float* __restrict__ gh_buf, float* __restrict__ decS,
                             unsigned int* __restrict__ bar) {
  extern __shared__ char smem[];
  const int ROWS = 192, STR = 520;                 // 520 bf16: 16B-aligned rows
  __bf16* Wl  = (__bf16*)smem;                     // 192 x 520 bf16
  __bf16* hbf = (__bf16*)(smem + ROWS * STR * 2);  // 512 bf16 (32B aligned)
  float*  hfl = (float*)(smem + ROWS * STR * 2 + H_ * 2);
  const int row0 = blockIdx.x * ROWS;

  for (int i = threadIdx.x; i < ROWS * H_; i += blockDim.x) {
    int r = i >> 9, c = i & (H_ - 1);
    Wl[r * STR + c] = (__bf16)Whd[(size_t)(row0 + r) * H_ + c];
  }
  for (int j = threadIdx.x; j < H_; j += blockDim.x) {
    float h = h0[j];
    hfl[j] = h; hbf[j] = (__bf16)h;
  }
  __syncthreads();

  const int lane  = threadIdx.x & 31;
  const int wv    = threadIdx.x >> 5;   // 12 waves, one 16-row tile each
  const int mbase = wv * 16;
  const v8f vz = {0.f, 0.f, 0.f, 0.f, 0.f, 0.f, 0.f, 0.f};
  const __bf16* arow = Wl + (mbase + (lane & 15)) * STR + ((lane >> 4) << 4);
  const __bf16* bcol = hbf + ((lane >> 4) << 4);

  for (int t = 0; t < TD_; ++t) {
    float* ghw = gh_buf + (t & 1) * (3 * H_);
    v8f acc0 = vz, acc1 = vz;
#pragma unroll
    for (int k0 = 0; k0 < H_; k0 += 64) {
      {
        const __bf16* ap = arow + k0;
        v8bf lo = *(const v8bf*)ap;
        v8bf hi = *(const v8bf*)(ap + 8);
        v16bf a;
#pragma unroll
        for (int e = 0; e < 8; ++e) { a[e] = lo[e]; a[e + 8] = hi[e]; }
        v16bf b = *(const v16bf*)(bcol + k0);            // h bcast to 16 cols
        acc0 = wmma_bf16(a, b, acc0);
      }
      {
        const __bf16* ap = arow + k0 + 32;
        v8bf lo = *(const v8bf*)ap;
        v8bf hi = *(const v8bf*)(ap + 8);
        v16bf a;
#pragma unroll
        for (int e = 0; e < 8; ++e) { a[e] = lo[e]; a[e + 8] = hi[e]; }
        v16bf b = *(const v16bf*)(bcol + k0 + 32);
        acc1 = wmma_bf16(a, b, acc1);
      }
    }
    if ((lane & 15) == 0) {                         // column n==0 holds the matvec
      int base = row0 + mbase + ((lane >> 4) << 3);
#pragma unroll
      for (int i = 0; i < 8; ++i) ghw[base + i] = acc0[i] + acc1[i] + bhd[base + i];
    }
    __threadfence();
    __syncthreads();
    if (threadIdx.x == 0) {
      __hip_atomic_fetch_add(bar, 1u, __ATOMIC_RELEASE, __HIP_MEMORY_SCOPE_AGENT);
      const unsigned int tgt = 8u * (unsigned int)(t + 1);
      while (__hip_atomic_load(bar, __ATOMIC_ACQUIRE, __HIP_MEMORY_SCOPE_AGENT) < tgt)
        __builtin_amdgcn_s_sleep(1);
    }
    __syncthreads();
    __threadfence();
    const float* gi = GI + (size_t)t * (3 * H_);
    for (int j = threadIdx.x; j < H_; j += blockDim.x) {  // all blocks redundantly
      float rr = 1.f / (1.f + __expf(-(gi[j] + ghw[j])));
      float zz = 1.f / (1.f + __expf(-(gi[H_ + j] + ghw[H_ + j])));
      float nn = tanhf(gi[2 * H_ + j] + rr * ghw[2 * H_ + j]);
      float hn = (1.f - zz) * nn + zz * hfl[j];
      hfl[j] = hn; hbf[j] = (__bf16)hn;
      if (blockIdx.x == 0) decS[(size_t)t * H_ + j] = hn;
    }
    __syncthreads();
  }
}

// ---- fused additive attention + softmax + context + concat(bf16) ------------
__global__ void attn_ctx_kernel(const float* __restrict__ encP, const float* __restrict__ decP,
                                const float* __restrict__ vat, const float* __restrict__ encH,
                                const float* __restrict__ decS, __bf16* __restrict__ catB) {
  __shared__ float sc[TE_];
  __shared__ float dp[A_];
  __shared__ float invs;
  const int t = blockIdx.x;
  for (int a = threadIdx.x; a < A_; a += blockDim.x) dp[a] = decP[(size_t)t * A_ + a];
  __syncthreads();
  const int lane = threadIdx.x & 31, wv = threadIdx.x >> 5;  // 8 waves
  for (int e = wv; e < TE_; e += 8) {
    const float* ep = encP + (size_t)e * A_;                 // encP includes b_attn
    float s = 0.f;
    for (int a = lane; a < A_; a += 32) s += tanhf(ep[a] + dp[a]) * vat[a];
#pragma unroll
    for (int off = 16; off > 0; off >>= 1) s += __shfl_down(s, off);
    if (lane == 0) sc[e] = s;
  }
  __syncthreads();
  if (wv == 0) {
    float m = -3.4e38f;
    for (int e = lane; e < TE_; e += 32) m = fmaxf(m, sc[e]);
#pragma unroll
    for (int off = 16; off > 0; off >>= 1) m = fmaxf(m, __shfl_xor(m, off));
    float sum = 0.f;
    for (int e = lane; e < TE_; e += 32) { float v = __expf(sc[e] - m); sc[e] = v; sum += v; }
#pragma unroll
    for (int off = 16; off > 0; off >>= 1) sum += __shfl_xor(sum, off);
    if (lane == 0) invs = 1.f / sum;
  }
  __syncthreads();
  const float inv = invs;
  for (int h = threadIdx.x; h < H_; h += blockDim.x) {
    float c = 0.f;
    for (int e = 0; e < TE_; ++e) c += sc[e] * encH[(size_t)e * H_ + h];
    c *= inv;
    catB[(size_t)t * 1024 + h]      = (__bf16)c;
    catB[(size_t)t * 1024 + H_ + h] = (__bf16)decS[(size_t)t * H_ + h];
  }
}

extern "C" void kernel_launch(void* const* d_in, const int* in_sizes, int n_in,
                              void* d_out, int out_size, void* d_ws, size_t ws_size,
                              hipStream_t stream) {
  const float* x     = (const float*)d_in[0];
  const float* y     = (const float*)d_in[1];
  const float* Wi_e  = (const float*)d_in[2];
  // d_in[3] = Wh_e: unused (encoder runs from zero hidden, gh = bh_e)
  const float* bi_e  = (const float*)d_in[4];
  const float* bh_e  = (const float*)d_in[5];
  const float* Wi_d  = (const float*)d_in[6];
  const float* Wh_d  = (const float*)d_in[7];
  const float* bi_d  = (const float*)d_in[8];
  const float* bh_d  = (const float*)d_in[9];
  const float* Wh_at = (const float*)d_in[10];
  const float* Ws_at = (const float*)d_in[11];
  const float* b_at  = (const float*)d_in[12];
  const float* v_at  = (const float*)d_in[13];
  const float* W1    = (const float*)d_in[14];
  const float* b1    = (const float*)d_in[15];
  const float* W2    = (const float*)d_in[16];
  const float* b2    = (const float*)d_in[17];
  float* out = (float*)d_out;

  char* w = (char*)d_ws;
  auto alloc = [&](size_t bytes) { char* p = w; w += (bytes + 255) & ~size_t(255); return p; };
  float*        GI_e = (float*)alloc((size_t)TE_ * 3 * H_ * 4);
  float*        GI_d = (float*)alloc((size_t)TD_ * 3 * H_ * 4);
  float*        encH = (float*)alloc((size_t)TE_ * H_ * 4);
  float*        encP = (float*)alloc((size_t)TE_ * A_ * 4);
  float*        decP = (float*)alloc((size_t)TD_ * A_ * 4);
  float*        decS = (float*)alloc((size_t)TD_ * H_ * 4);
  float*        hcur = (float*)alloc((size_t)H_ * 4);
  float*        ghb  = (float*)alloc((size_t)2 * 3 * H_ * 4);
  unsigned int* bar  = (unsigned int*)alloc(256);
  __bf16*       catB = (__bf16*)alloc((size_t)TD_ * 2 * H_ * 2);
  __bf16*       o1B  = (__bf16*)alloc((size_t)TD_ * 2 * V_ * 2);

  init_bar_kernel<<<1, 1, 0, stream>>>(bar);
  // encoder gate pre-activations: GI_e = X @ Wi_e^T + bi_e   [256,1536]
  gemm_bt<float, float><<<dim3(6, 2), 256, 0, stream>>>(x, Wi_e, bi_e, GI_e,
                                                        3 * H_, D_, D_, D_, 3 * H_);
  // decoder input pre-activations: GI_d = Y @ Wi_d^T + bi_d  [128,1536]
  gemm_bt<float, float><<<dim3(6, 1), 256, 0, stream>>>(y, Wi_d, bi_d, GI_d,
                                                        3 * H_, D_, D_, D_, 3 * H_);
  enc_gates_kernel<<<(TE_ * H_) / 256, 256, 0, stream>>>(GI_e, bh_e, encH, hcur);
  // encP = enc_H @ Wh_attn^T + b_attn  [256,256]
  gemm_bt<float, float><<<dim3(1, 2), 256, 0, stream>>>(encH, Wh_at, b_at, encP,
                                                        A_, H_, H_, H_, A_);
  const int dec_lds = 192 * 520 * 2 + H_ * 2 + H_ * 4;   // 202752 B (<320KB/WGP)
  decoder_scan<<<8, 384, dec_lds, stream>>>(Wh_d, bh_d, GI_d, hcur, ghb, decS, bar);
  // decP = dec_S @ Ws_attn^T  [128,256]
  gemm_bt<float, float><<<dim3(1, 1), 256, 0, stream>>>(decS, Ws_at, nullptr, decP,
                                                        A_, H_, H_, H_, A_);
  attn_ctx_kernel<<<TD_, 256, 0, stream>>>(encP, decP, v_at, encH, decS, catB);
  // out1 = concat @ W1^T + b1  [128,20000]  (bf16 A in, bf16 out)
  gemm_bt<__bf16, __bf16><<<dim3(79, 1), 256, 0, stream>>>(catB, W1, b1, o1B,
                                                           2 * V_, 2 * H_, 2 * H_, 2 * H_, 2 * V_);
  // out = out1 @ W2^T + b2  [128,10000]  (the 51 GFLOP / 800MB stream)
  gemm_bt<__bf16, float><<<dim3(40, 1), 256, 0, stream>>>(o1B, W2, b2, out,
                                                          V_, 2 * V_, 2 * V_, 2 * V_, V_);
}
